// TransformerDecoderLayer_40570261078271
// MI455X (gfx1250) — matrix operand order repaired and checked by reading, not compiled, not run on hardware
//
#include <hip/hip_runtime.h>
#include <hip/hip_bf16.h>

typedef __bf16 bf16;
typedef __attribute__((ext_vector_type(16))) __bf16 v16bf;
typedef __attribute__((ext_vector_type(8)))  __bf16 v8bf;
typedef __attribute__((ext_vector_type(8)))  float  v8f;

#define BS   8
#define QN   300
#define NQ   2400        // BS*QN
#define CH   256
#define HW   400
#define OUTS 7
#define CELLS 49

// ---------------------------------------------------------------------------
// valid-ratio per batch from padding mask (True == padded)
// ---------------------------------------------------------------------------
__global__ void vr_kernel(const int* __restrict__ mask, float* __restrict__ vr) {
    int b = threadIdx.x;
    if (b >= BS) return;
    int width = 0, height = 0;
    for (int w = 0; w < 20; ++w) {
        int any = 0;
        for (int h = 0; h < 20; ++h) any |= (mask[b * HW + h * 20 + w] == 0);
        width += any;
    }
    for (int h = 0; h < 20; ++h) {
        int any = 0;
        for (int w = 0; w < 20; ++w) any |= (mask[b * HW + h * 20 + w] == 0);
        height += any;
    }
    vr[b * 2 + 0] = (float)width  / 20.0f;
    vr[b * 2 + 1] = (float)height / 20.0f;
}

// ---------------------------------------------------------------------------
// weight prep: fp32 (din x dout) -> bf16 transposed (doutPad x din), bias pad
// ---------------------------------------------------------------------------
__global__ void prep_weight(const float* __restrict__ w, bf16* __restrict__ wt,
                            const float* __restrict__ b, float* __restrict__ bp,
                            int din, int dout, int doutPad) {
    int i = blockIdx.x * blockDim.x + threadIdx.x;
    int total = doutPad * din;
    if (i < total) {
        int o = i / din;
        int k = i - o * din;
        wt[(size_t)o * din + k] = (o < dout) ? (bf16)w[(size_t)k * dout + o] : (bf16)0.0f;
    }
    if (i < doutPad) bp[i] = (i < dout) ? b[i] : 0.0f;
}

// ---------------------------------------------------------------------------
// ROI align: memory (400,8,256) f32 -> qc (2400, 49, 256) bf16
// one block per query, one thread per channel (coalesced channel loads)
// ---------------------------------------------------------------------------
__global__ __launch_bounds__(256) void roi_kernel(
    const float* __restrict__ memory, const float* __restrict__ boxes,
    const float* __restrict__ vr, bf16* __restrict__ qc) {
    int n = blockIdx.x;
    int c = threadIdx.x;
    int b = n / QN;
    const float* box = boxes + (size_t)n * 4;
    float cx = box[0], cy = box[1], hw2 = box[2] * 0.5f, hh2 = box[3] * 0.5f;
    float sx = 20.0f * vr[b * 2 + 0];
    float sy = 20.0f * vr[b * 2 + 1];
    float X1 = (cx - hw2) * sx, Y1 = (cy - hh2) * sy;
    float X2 = (cx + hw2) * sx, Y2 = (cy + hh2) * sy;
    float x1 = X1 - 0.5f, y1 = Y1 - 0.5f;
    float bw = (X2 - X1) / 7.0f, bh = (Y2 - Y1) / 7.0f;
    const float* fb = memory + b * CH + c;   // feat(b,c,y,x) = fb[(y*20+x)*BS*CH]
    for (int oy = 0; oy < OUTS; ++oy) {
        for (int ox = 0; ox < OUTS; ++ox) {
            float acc = 0.0f;
#pragma unroll
            for (int s = 0; s < 4; ++s) {
                int sy_ = s >> 1, sx_ = s & 1;
                float gy = ((float)(oy * 2 + sy_) + 0.5f) * 0.5f;
                float gx = ((float)(ox * 2 + sx_) + 0.5f) * 0.5f;
                float y = y1 + gy * bh;
                float x = x1 + gx * bw;
                float valid = (y > -1.0f && y < 20.0f && x > -1.0f && x < 20.0f) ? 1.0f : 0.0f;
                y = fminf(fmaxf(y, 0.0f), 19.0f);
                x = fminf(fmaxf(x, 0.0f), 19.0f);
                int y0 = (int)floorf(y), x0 = (int)floorf(x);
                int y1i = min(y0 + 1, 19), x1i = min(x0 + 1, 19);
                float ly = y - (float)y0, lx = x - (float)x0;
                float hy = 1.0f - ly, hx = 1.0f - lx;
                float v00 = fb[(y0 * 20 + x0) * (BS * CH)];
                float v01 = fb[(y0 * 20 + x1i) * (BS * CH)];
                float v10 = fb[(y1i * 20 + x0) * (BS * CH)];
                float v11 = fb[(y1i * 20 + x1i) * (BS * CH)];
                acc += (v00 * hy * hx + v01 * hy * lx + v10 * ly * hx + v11 * ly * lx) * valid;
            }
            qc[((size_t)n * CELLS + oy * 7 + ox) * CH + c] = (bf16)(acc * 0.25f);
        }
    }
}

// ---------------------------------------------------------------------------
// register-blocked bf16 WMMA GEMM: D = act(A(MxK) * Wt^T + bias)
//   A row-major bf16 (ld = K), Wt is (Npad64 x K) bf16 (pre-transposed weights)
//   Each wave computes a 32(M) x 64(N) block: 2x4 = 8 WMMA accumulators,
//   reusing every A fragment 4x and every B fragment 2x (21 FLOP/L2-byte vs 8
//   for the naive wave-per-tile version). 4 waves/block cover 128 rows.
//   outMode: 0 = bf16 row-major (ld=N), 1 = bf16 conv-permuted (n,o,hw),
//            2 = f32 row-major (ld=N)
//   act: 0 none, 1 relu, 2 tanh
// ---------------------------------------------------------------------------
__global__ __launch_bounds__(128) void wgemm_bf16(
    const bf16* __restrict__ A, const bf16* __restrict__ Wt,
    const float* __restrict__ bias, void* __restrict__ outp,
    int M, int N, int K, int act, int outMode) {
    const int lane  = threadIdx.x & 31;
    const int wave  = threadIdx.x >> 5;
    const int mTile = blockIdx.x * 4 + wave;     // 32-row tiles
    if (mTile * 32 >= M) return;                 // uniform per wave: EXEC stays all-1s
    const int mBase = mTile * 32;
    const int nBase = blockIdx.y * 64;

    const int l15   = lane & 15;
    const int hi    = lane >> 4;                 // 0 or 1
    const int aKoff = hi * 16;                   // A frag: hi lanes hold K+16..K+31
    const int bKoff = hi * 16;                   // B frag: hi lanes hold K+16..K+31

    const bf16* aRow0 = A + (size_t)(mBase + l15) * K;
    const bf16* aRow1 = aRow0 + (size_t)16 * K;
    const bf16* bRow0 = Wt + (size_t)(nBase + l15) * K;
    const bf16* bRow1 = bRow0 + (size_t)16 * K;
    const bf16* bRow2 = bRow0 + (size_t)32 * K;
    const bf16* bRow3 = bRow0 + (size_t)48 * K;

    v8f acc00 = {}, acc01 = {}, acc02 = {}, acc03 = {};
    v8f acc10 = {}, acc11 = {}, acc12 = {}, acc13 = {};

    for (int kb = 0; kb < K; kb += 32) {
        // prefetch next K block (global_prefetch_b8)
        if (kb + 32 < K) {
            __builtin_prefetch(aRow0 + kb + 32, 0, 1);
            __builtin_prefetch(bRow0 + kb + 32, 0, 1);
        }
        v8bf a00 = *(const v8bf*)(aRow0 + kb + aKoff);
        v8bf a01 = *(const v8bf*)(aRow0 + kb + 8 + aKoff);
        v16bf a0 = __builtin_shufflevector(a00, a01, 0, 1, 2, 3, 4, 5, 6, 7,
                                                     8, 9, 10, 11, 12, 13, 14, 15);
        v8bf a10 = *(const v8bf*)(aRow1 + kb + aKoff);
        v8bf a11 = *(const v8bf*)(aRow1 + kb + 8 + aKoff);
        v16bf a1 = __builtin_shufflevector(a10, a11, 0, 1, 2, 3, 4, 5, 6, 7,
                                                     8, 9, 10, 11, 12, 13, 14, 15);
        v16bf b0 = *(const v16bf*)(bRow0 + kb + bKoff);
        v16bf b1 = *(const v16bf*)(bRow1 + kb + bKoff);
        v16bf b2 = *(const v16bf*)(bRow2 + kb + bKoff);
        v16bf b3 = *(const v16bf*)(bRow3 + kb + bKoff);
        acc00 = __builtin_amdgcn_wmma_f32_16x16x32_bf16(false, a0, false, b0, (short)0, acc00, false, false);
        acc01 = __builtin_amdgcn_wmma_f32_16x16x32_bf16(false, a0, false, b1, (short)0, acc01, false, false);
        acc02 = __builtin_amdgcn_wmma_f32_16x16x32_bf16(false, a0, false, b2, (short)0, acc02, false, false);
        acc03 = __builtin_amdgcn_wmma_f32_16x16x32_bf16(false, a0, false, b3, (short)0, acc03, false, false);
        acc10 = __builtin_amdgcn_wmma_f32_16x16x32_bf16(false, a1, false, b0, (short)0, acc10, false, false);
        acc11 = __builtin_amdgcn_wmma_f32_16x16x32_bf16(false, a1, false, b1, (short)0, acc11, false, false);
        acc12 = __builtin_amdgcn_wmma_f32_16x16x32_bf16(false, a1, false, b2, (short)0, acc12, false, false);
        acc13 = __builtin_amdgcn_wmma_f32_16x16x32_bf16(false, a1, false, b3, (short)0, acc13, false, false);
    }

    const int mOff = hi * 8;                     // C/D frag: hi lanes hold M+8
    v8f accs[2][4] = {{acc00, acc01, acc02, acc03}, {acc10, acc11, acc12, acc13}};
#pragma unroll
    for (int j = 0; j < 4; ++j) {
        const int gn = nBase + j * 16 + l15;
        if (gn >= N) continue;
        const float bv = bias[gn];
#pragma unroll
        for (int i = 0; i < 2; ++i) {
#pragma unroll
            for (int v = 0; v < 8; ++v) {
                int gm = mBase + i * 16 + mOff + v;
                float r = accs[i][j][v] + bv;
                if (act == 1) r = fmaxf(r, 0.0f);
                else if (act == 2) r = tanhf(r);
                if (outMode == 0) {
                    ((bf16*)outp)[(size_t)gm * N + gn] = (bf16)r;
                } else if (outMode == 1) {       // pt[(n, o, hw)] from m = n*49+hw
                    int nq = gm / CELLS;
                    int hw = gm - nq * CELLS;
                    ((bf16*)outp)[(size_t)nq * (64 * CELLS) + gn * CELLS + hw] = (bf16)r;
                } else {
                    ((float*)outp)[(size_t)gm * N + gn] = r;
                }
            }
        }
    }
}

// ---------------------------------------------------------------------------
// grid_sample: qc (2400,49,256) bf16 + points (2400,16) f32 -> feat (2400,2048) bf16
// feat col layout = c*8 + p  (matches sampled.reshape(bs, nq, c*8))
// ---------------------------------------------------------------------------
__global__ __launch_bounds__(256) void sample_kernel(
    const bf16* __restrict__ qc, const float* __restrict__ points,
    bf16* __restrict__ feat) {
    int n = blockIdx.x;
    int c = threadIdx.x;
    const bf16* q = qc + (size_t)n * CELLS * CH + c;
#pragma unroll
    for (int p = 0; p < 8; ++p) {
        float px = points[n * 16 + p * 2 + 0];
        float py = points[n * 16 + p * 2 + 1];
        float ix = (px + 1.0f) * 3.5f - 0.5f;
        float iy = (py + 1.0f) * 3.5f - 0.5f;
        int x0 = (int)floorf(ix), y0 = (int)floorf(iy);
        float lx = ix - (float)x0, ly = iy - (float)y0;
        float v00 = 0.0f, v01 = 0.0f, v10 = 0.0f, v11 = 0.0f;
        if (y0 >= 0 && y0 < 7) {
            if (x0 >= 0 && x0 < 7)         v00 = (float)q[(y0 * 7 + x0) * CH];
            if (x0 + 1 >= 0 && x0 + 1 < 7) v01 = (float)q[(y0 * 7 + x0 + 1) * CH];
        }
        if (y0 + 1 >= 0 && y0 + 1 < 7) {
            if (x0 >= 0 && x0 < 7)         v10 = (float)q[((y0 + 1) * 7 + x0) * CH];
            if (x0 + 1 >= 0 && x0 + 1 < 7) v11 = (float)q[((y0 + 1) * 7 + x0 + 1) * CH];
        }
        float r = v00 * (1.0f - ly) * (1.0f - lx) + v01 * (1.0f - ly) * lx
                + v10 * ly * (1.0f - lx) + v11 * ly * lx;
        feat[(size_t)n * 2048 + c * 8 + p] = (bf16)r;
    }
}

// ---------------------------------------------------------------------------
// host side
// ---------------------------------------------------------------------------
static inline char* bump(char*& p, size_t bytes) {
    char* r = p;
    p += (bytes + 255) & ~(size_t)255;
    return r;
}

static void launch_gemm(const bf16* A, const bf16* Wt, const float* bp, void* out,
                        int M, int N, int K, int act, int outMode, hipStream_t s) {
    int Npad = (N + 63) & ~63;
    dim3 grid((M / 32 + 3) / 4, Npad / 64);
    wgemm_bf16<<<grid, 128, 0, s>>>(A, Wt, bp, out, M, N, K, act, outMode);
}

static void launch_prep(const float* w, bf16* wt, const float* b, float* bp,
                        int din, int dout, hipStream_t s) {
    int doutPad = (dout + 63) & ~63;
    int total = doutPad * din;
    prep_weight<<<(total + 255) / 256, 256, 0, s>>>(w, wt, b, bp, din, dout, doutPad);
}

extern "C" void kernel_launch(void* const* d_in, const int* in_sizes, int n_in,
                              void* d_out, int out_size, void* d_ws, size_t ws_size,
                              hipStream_t stream) {
    (void)in_sizes; (void)n_in; (void)out_size; (void)ws_size;
    const float* memory = (const float*)d_in[1];
    const int*   mask   = (const int*)d_in[2];
    const float* boxes  = (const float*)d_in[3];
    const float* conv_w = (const float*)d_in[6];
    const float* conv_b = (const float*)d_in[7];
    const float* p2_w1  = (const float*)d_in[8];
    const float* p2_b1  = (const float*)d_in[9];
    const float* p2_w2  = (const float*)d_in[10];
    const float* p2_b2  = (const float*)d_in[11];
    const float* p2_w3  = (const float*)d_in[12];
    const float* p2_b3  = (const float*)d_in[13];
    const float* p2_w4  = (const float*)d_in[14];
    const float* p2_b4  = (const float*)d_in[15];
    const float* mlp_w1 = (const float*)d_in[16];
    const float* mlp_b1 = (const float*)d_in[17];
    const float* mlp_w2 = (const float*)d_in[18];
    const float* mlp_b2 = (const float*)d_in[19];
    const float* mlp_w3 = (const float*)d_in[20];
    const float* mlp_b3 = (const float*)d_in[21];
    const float* mlp_w4 = (const float*)d_in[22];
    const float* mlp_b4 = (const float*)d_in[23];

    char* p = (char*)d_ws;
    float* vr      = (float*)bump(p, BS * 2 * sizeof(float));
    bf16* qc       = (bf16*)bump(p, (size_t)NQ * CELLS * CH * 2);          // 60 MB
    bf16* pt       = (bf16*)bump(p, (size_t)NQ * 3136 * 2);                // 15 MB
    bf16* x1       = (bf16*)bump(p, (size_t)NQ * 256 * 2);
    bf16* x2       = (bf16*)bump(p, (size_t)NQ * 512 * 2);
    bf16* x3       = (bf16*)bump(p, (size_t)NQ * 512 * 2);
    float* pts     = (float*)bump(p, (size_t)NQ * 16 * 4);
    bf16* feat     = (bf16*)bump(p, (size_t)NQ * 2048 * 2);
    bf16* y1       = (bf16*)bump(p, (size_t)NQ * 1024 * 2);
    bf16* y2       = (bf16*)bump(p, (size_t)NQ * 512 * 2);
    bf16* y3       = (bf16*)bump(p, (size_t)NQ * 256 * 2);
    bf16* conv_wT  = (bf16*)bump(p, (size_t)64 * 256 * 2);
    bf16* p2_w1T   = (bf16*)bump(p, (size_t)256 * 3136 * 2);
    bf16* p2_w2T   = (bf16*)bump(p, (size_t)512 * 256 * 2);
    bf16* p2_w3T   = (bf16*)bump(p, (size_t)512 * 512 * 2);
    bf16* p2_w4T   = (bf16*)bump(p, (size_t)64 * 512 * 2);                 // padded to 64 rows
    bf16* mlp_w1T  = (bf16*)bump(p, (size_t)1024 * 2048 * 2);
    bf16* mlp_w2T  = (bf16*)bump(p, (size_t)512 * 1024 * 2);
    bf16* mlp_w3T  = (bf16*)bump(p, (size_t)256 * 512 * 2);
    bf16* mlp_w4T  = (bf16*)bump(p, (size_t)64 * 256 * 2);                 // padded to 64 rows
    float* conv_bP = (float*)bump(p, 64 * 4);
    float* p2_b1P  = (float*)bump(p, 256 * 4);
    float* p2_b2P  = (float*)bump(p, 512 * 4);
    float* p2_b3P  = (float*)bump(p, 512 * 4);
    float* p2_b4P  = (float*)bump(p, 64 * 4);
    float* mlp_b1P = (float*)bump(p, 1024 * 4);
    float* mlp_b2P = (float*)bump(p, 512 * 4);
    float* mlp_b3P = (float*)bump(p, 512 * 4);
    float* mlp_b4P = (float*)bump(p, 64 * 4);

    // weight prep (bf16 + transpose so B fragments are contiguous in K)
    launch_prep(conv_w, conv_wT, conv_b, conv_bP, 256, 64, stream);
    launch_prep(p2_w1,  p2_w1T,  p2_b1,  p2_b1P,  3136, 256, stream);
    launch_prep(p2_w2,  p2_w2T,  p2_b2,  p2_b2P,  256, 512, stream);
    launch_prep(p2_w3,  p2_w3T,  p2_b3,  p2_b3P,  512, 512, stream);
    launch_prep(p2_w4,  p2_w4T,  p2_b4,  p2_b4P,  512, 16, stream);
    launch_prep(mlp_w1, mlp_w1T, mlp_b1, mlp_b1P, 2048, 1024, stream);
    launch_prep(mlp_w2, mlp_w2T, mlp_b2, mlp_b2P, 1024, 512, stream);
    launch_prep(mlp_w3, mlp_w3T, mlp_b3, mlp_b3P, 512, 256, stream);
    launch_prep(mlp_w4, mlp_w4T, mlp_b4, mlp_b4P, 256, 1, stream);

    vr_kernel<<<1, 32, 0, stream>>>(mask, vr);
    roi_kernel<<<NQ, 256, 0, stream>>>(memory, boxes, vr, qc);

    // conv1x1 as GEMM (117600x256)x(256x64), permuted store to (n, o, hw)
    launch_gemm(qc, conv_wT, conv_bP, pt, NQ * CELLS, 64, 256, 1, 1, stream);
    // p2 MLP
    launch_gemm(pt, p2_w1T, p2_b1P, x1, NQ, 256, 3136, 1, 0, stream);
    launch_gemm(x1, p2_w2T, p2_b2P, x2, NQ, 512, 256, 1, 0, stream);
    launch_gemm(x2, p2_w3T, p2_b3P, x3, NQ, 512, 512, 1, 0, stream);
    launch_gemm(x3, p2_w4T, p2_b4P, pts, NQ, 16, 512, 2, 2, stream);  // tanh, f32
    // sample qc at the 8 predicted points
    sample_kernel<<<NQ, 256, 0, stream>>>(qc, pts, feat);
    // final MLP
    launch_gemm(feat, mlp_w1T, mlp_b1P, y1, NQ, 1024, 2048, 1, 0, stream);
    launch_gemm(y1, mlp_w2T, mlp_b2P, y2, NQ, 512, 1024, 1, 0, stream);
    launch_gemm(y2, mlp_w3T, mlp_b3P, y3, NQ, 256, 512, 1, 0, stream);
    launch_gemm(y3, mlp_w4T, mlp_b4P, d_out, NQ, 1, 256, 0, 2, stream);
}
